// GraphTransformerModel_644245094994
// MI455X (gfx1250) — matrix operand order repaired and checked by reading, not compiled, not run on hardware
//
#include <hip/hip_runtime.h>
#include <cstdint>

typedef __attribute__((ext_vector_type(16))) _Float16 v16h;
typedef __attribute__((ext_vector_type(4)))  _Float16 v4h;
typedef __attribute__((ext_vector_type(8)))  float    v8f;

// ---------------------------------------------------------------------------
// sign-aware float atomic max (pure int atomics, no CAS loop)
// ---------------------------------------------------------------------------
__device__ __forceinline__ void atomicMaxFloat(float* addr, float val) {
    if (val >= 0.0f) {
        atomicMax((int*)addr, __float_as_int(val));
    } else {
        atomicMin((unsigned int*)addr, __float_as_uint(val));
    }
}

// ---------------------------------------------------------------------------
// Fused 4-projection GEMM:
//   Yw[n_rows x NCOLS] = op(X)[n_rows x 64] * Ww[64 x NCOLS] + bw,  w = 0..3
// op = relu if RELU.  Block = 128 threads = 4 waves; wave owns a 16-row tile.
// LDS holds A and B tiles pre-swizzled into WMMA fragment order, so each
// fragment is one 32-byte LDS load (2x ds_load_b128) -- no b16 repacking.
//
// Fragment addressing (v_wmma_f32_16x16x32_f16, wave32):
//   A 16x32 f16 : lane = half*16 + m, elem j -> k = half*16 + j
//   B 32x16 f16 : lane = half*16 + n, elem j -> k = half*16 + j
//   C/D 16x16 f32: vgpr r, lane half -> row = half*8 + r, col = lane&15
// ---------------------------------------------------------------------------
template <int NCOLS, int RELU>
__global__ __launch_bounds__(128)
void gemm_proj4_kernel(const float* __restrict__ X,
                       const float* __restrict__ W0, const float* __restrict__ W1,
                       const float* __restrict__ W2, const float* __restrict__ W3,
                       const float* __restrict__ pb0, const float* __restrict__ pb1,
                       const float* __restrict__ pb2, const float* __restrict__ pb3,
                       float* __restrict__ Y0, float* __restrict__ Y1,
                       float* __restrict__ Y2, float* __restrict__ Y3,
                       int n_rows)
{
    constexpr int NT = NCOLS / 16;                 // 16-col output tiles
    __shared__ _Float16 ldsB[4][NT][32][32];       // [w][nt][lane][kb*16+j]
    __shared__ _Float16 ldsA[4][32][32];           // [wave][lane][kb*16+j]

    const int tid  = threadIdx.x;
    const int wave = tid >> 5;
    const int lane = tid & 31;

    // ---- cooperative load of the 4 weight matrices, swizzled to B-frag order
    const float* Wp[4] = {W0, W1, W2, W3};
#pragma unroll
    for (int w = 0; w < 4; ++w) {
        const float* Wg = Wp[w];
        for (int idx = tid; idx < 64 * NCOLS; idx += 128) {
            const int k  = idx / NCOLS;            // coalesced over n
            const int n  = idx % NCOLS;
            const int nt = n >> 4, nl = n & 15;
            const int kb = k >> 5, kl = k & 31;
            const int hf = kl >> 4, j = kl & 15;
            ldsB[w][nt][hf * 16 + nl][kb * 16 + j] = (_Float16)Wg[idx];
        }
    }

    // ---- per-wave X tile load, swizzled to A-frag order (v4h stores) ----
    const int rowBase = blockIdx.x * 64 + wave * 16;
    if (rowBase < n_rows) {
        for (int i = lane; i < 256; i += 32) {     // 16 rows x 16 float4
            const int r  = i >> 4;
            const int c4 = i & 15;
            float4 f = ((const float4*)(X + (size_t)(rowBase + r) * 64))[c4];
            if (RELU) {
                f.x = fmaxf(f.x, 0.0f); f.y = fmaxf(f.y, 0.0f);
                f.z = fmaxf(f.z, 0.0f); f.w = fmaxf(f.w, 0.0f);
            }
            const int k  = c4 * 4;
            const int kb = k >> 5, kl = k & 31;
            const int hf = kl >> 4;
            const int j0 = kl & 15;   // multiple of 4
            v4h hv;
            hv[0] = (_Float16)f.x; hv[1] = (_Float16)f.y;
            hv[2] = (_Float16)f.z; hv[3] = (_Float16)f.w;
            *(v4h*)&ldsA[wave][hf * 16 + r][kb * 16 + j0] = hv;
        }
    }
    __syncthreads();
    if (rowBase >= n_rows) return;

    const int half = lane >> 4;
    const int m    = lane & 15;

    // ---- A fragments: single 32B LDS loads ----
    const v16h a0 = *(const v16h*)&ldsA[wave][lane][0];
    const v16h a1 = *(const v16h*)&ldsA[wave][lane][16];

    const float* bp[4] = {pb0, pb1, pb2, pb3};
    float*       Yp[4] = {Y0, Y1, Y2, Y3};

#pragma unroll
    for (int w = 0; w < 4; ++w) {
#pragma unroll
        for (int nt = 0; nt < NT; ++nt) {
            const v16h bf0 = *(const v16h*)&ldsB[w][nt][lane][0];
            const v16h bf1 = *(const v16h*)&ldsB[w][nt][lane][16];
            v8f c = {};
            c = __builtin_amdgcn_wmma_f32_16x16x32_f16(false, a0, false, bf0,
                                                       (short)0, c, false, false);
            c = __builtin_amdgcn_wmma_f32_16x16x32_f16(false, a1, false, bf1,
                                                       (short)0, c, false, false);
            const float bv = bp[w][nt * 16 + m];
            float* yrow = Yp[w] + (size_t)(rowBase + half * 8) * NCOLS + nt * 16 + m;
#pragma unroll
            for (int r = 0; r < 8; ++r) {
                yrow[(size_t)r * NCOLS] = c[r] + bv;
            }
        }
    }
}

// ---------------------------------------------------------------------------
// init amax = -FLT_MAX, den = 0
// ---------------------------------------------------------------------------
__global__ void init_softmax_kernel(float* __restrict__ amax,
                                    float* __restrict__ den, int n)
{
    const int i = blockIdx.x * blockDim.x + threadIdx.x;
    if (i < n) { amax[i] = -3.402823466e38f; den[i] = 0.0f; }
}

// ---------------------------------------------------------------------------
// Pass 1: per-edge attention logits + segment max
// ---------------------------------------------------------------------------
__global__ void edge_logits_kernel(const float* __restrict__ q,
                                   const float* __restrict__ kmat,
                                   const int* __restrict__ src,
                                   const int* __restrict__ dst,
                                   float* __restrict__ alpha,
                                   float* __restrict__ amax,
                                   int E, int heads)
{
    const int e = blockIdx.x * blockDim.x + threadIdx.x;
    if (e >= E) return;
    const int s = src[e];
    const int d = dst[e];
    const int rl = heads * 32;
    const float* qr = q + (size_t)d * rl;
    const float* kr = kmat + (size_t)s * rl;
    const float scale = 0.17677669529663687f;   // 1/sqrt(32)
    for (int h = 0; h < heads; ++h) {
        float acc = 0.0f;
#pragma unroll
        for (int i = 0; i < 32; i += 4) {
            const float4 qv = *(const float4*)(qr + h * 32 + i);
            const float4 kv = *(const float4*)(kr + h * 32 + i);
            acc += qv.x * kv.x + qv.y * kv.y + qv.z * kv.z + qv.w * kv.w;
        }
        acc *= scale;
        alpha[(size_t)e * heads + h] = acc;
        atomicMaxFloat(&amax[d * heads + h], acc);
    }
}

// ---------------------------------------------------------------------------
// Pass 2: ex = exp(alpha - amax[dst]); den[dst] += ex (ex cached into alpha)
// ---------------------------------------------------------------------------
__global__ void edge_expsum_kernel(float* __restrict__ alpha,
                                   const int* __restrict__ dst,
                                   const float* __restrict__ amax,
                                   float* __restrict__ den,
                                   int E, int heads)
{
    const int e = blockIdx.x * blockDim.x + threadIdx.x;
    if (e >= E) return;
    const int d = dst[e];
    for (int h = 0; h < heads; ++h) {
        const float ex = __expf(alpha[(size_t)e * heads + h] - amax[d * heads + h]);
        alpha[(size_t)e * heads + h] = ex;
        atomicAdd(&den[d * heads + h], ex);
    }
}

// ---------------------------------------------------------------------------
// Pass 3: out[dst, c] += v[src, c] * ex[e, c>>5] / den[dst, c>>5]
// ---------------------------------------------------------------------------
__global__ void edge_aggregate_kernel(const float* __restrict__ ex,
                                      const float* __restrict__ den,
                                      const float* __restrict__ v,
                                      const int* __restrict__ src,
                                      const int* __restrict__ dst,
                                      float* __restrict__ out,
                                      int E, int heads, int log2C)
{
    const int C = heads * 32;
    const int t = blockIdx.x * blockDim.x + threadIdx.x;
    const int e = t >> log2C;
    const int c = t & (C - 1);
    if (e >= E) return;
    const int s = src[e];
    const int d = dst[e];
    const int h = c >> 5;
    const float coef = ex[(size_t)e * heads + h] / den[(size_t)d * heads + h];
    atomicAdd(&out[(size_t)d * C + c], v[(size_t)s * C + c] * coef);
}

// ---------------------------------------------------------------------------
// launcher
// ---------------------------------------------------------------------------
extern "C" void kernel_launch(void* const* d_in, const int* in_sizes, int n_in,
                              void* d_out, int out_size, void* d_ws, size_t ws_size,
                              hipStream_t stream)
{
    const float* x   = (const float*)d_in[0];
    const int*   ei  = (const int*)d_in[1];
    const float* Wq1 = (const float*)d_in[2];  const float* bq1 = (const float*)d_in[3];
    const float* Wk1 = (const float*)d_in[4];  const float* bk1 = (const float*)d_in[5];
    const float* Wv1 = (const float*)d_in[6];  const float* bv1 = (const float*)d_in[7];
    const float* Ws1 = (const float*)d_in[8];  const float* bs1 = (const float*)d_in[9];
    const float* Wq2 = (const float*)d_in[10]; const float* bq2 = (const float*)d_in[11];
    const float* Wk2 = (const float*)d_in[12]; const float* bk2 = (const float*)d_in[13];
    const float* Wv2 = (const float*)d_in[14]; const float* bv2 = (const float*)d_in[15];
    const float* Ws2 = (const float*)d_in[16]; const float* bs2 = (const float*)d_in[17];

    const int N = in_sizes[0] / 64;
    const int E = in_sizes[1] / 2;
    const int* src = ei;
    const int* dst = ei + E;

    // workspace layout (floats)
    float* ws = (float*)d_ws;
    const size_t N64 = (size_t)N * 64, N32 = (size_t)N * 32;
    float* q1    = ws;
    float* k1    = q1 + N64;
    float* v1    = k1 + N64;
    float* h     = v1 + N64;            // skip + aggregated messages (conv1 out)
    float* q2    = h + N64;
    float* k2    = q2 + N32;
    float* v2    = k2 + N32;
    float* alpha = v2 + N32;            // E*2 (conv1), reused E*1 (conv2)
    float* amax1 = alpha + (size_t)E * 2;
    float* den1  = amax1 + (size_t)N * 2;
    float* amax2 = den1 + (size_t)N * 2;
    float* den2  = amax2 + (size_t)N;
    float* out   = (float*)d_out;

    const int gemmBlocks = (N + 63) / 64;
    const int edgeBlocks = (E + 255) / 256;

    // ---------------- conv1: fused Q/K/V/skip projections ----------------
    gemm_proj4_kernel<64, 0><<<gemmBlocks, 128, 0, stream>>>(
        x, Wq1, Wk1, Wv1, Ws1, bq1, bk1, bv1, bs1, q1, k1, v1, h, N);

    init_softmax_kernel<<<(2 * N + 255) / 256, 256, 0, stream>>>(amax1, den1, 2 * N);
    edge_logits_kernel<<<edgeBlocks, 256, 0, stream>>>(q1, k1, src, dst, alpha, amax1, E, 2);
    edge_expsum_kernel<<<edgeBlocks, 256, 0, stream>>>(alpha, dst, amax1, den1, E, 2);
    {
        const long long tt = (long long)E * 64;
        edge_aggregate_kernel<<<(unsigned)((tt + 255) / 256), 256, 0, stream>>>(
            alpha, den1, v1, src, dst, h, E, 2, 6);
    }

    // ---------------- conv2: relu fused into the GEMM A-load ----------------
    gemm_proj4_kernel<32, 1><<<gemmBlocks, 128, 0, stream>>>(
        h, Wq2, Wk2, Wv2, Ws2, bq2, bk2, bv2, bs2, q2, k2, v2, out, N);

    init_softmax_kernel<<<(N + 255) / 256, 256, 0, stream>>>(amax2, den2, N);
    edge_logits_kernel<<<edgeBlocks, 256, 0, stream>>>(q2, k2, src, dst, alpha, amax2, E, 1);
    edge_expsum_kernel<<<edgeBlocks, 256, 0, stream>>>(alpha, dst, amax2, den2, E, 1);
    {
        const long long tt = (long long)E * 32;
        edge_aggregate_kernel<<<(unsigned)((tt + 255) / 256), 256, 0, stream>>>(
            alpha, den2, v2, src, dst, out, E, 1, 5);
    }
}
